// AdapterWithHyperNet_58652073394799
// MI455X (gfx1250) — matrix operand order repaired and checked by reading, not compiled, not run on hardware
//
#include <hip/hip_runtime.h>
#include <math.h>

// ---------------------------------------------------------------------------
// AdapterWithHyperNet for MI455X (gfx1250, wave32, WMMA + Tensor Data Mover).
//
//   t  = x @ Rd^T            GEMM  M=8192 N=64   K=4096   (wmma bf16)
//   t2 = H_b @ t             batched matvec (streams H, memory bound)
//   z  = sigmoid(t2@Ld^T+b1) GEMM  M=8192 N=1024 K=64
//   s  = z @ Ru^T            GEMM  M=8192 N=64   K=1024
//   s2 = H_b @ s             batched matvec
//   pre= s2 @ Lu^T + b2      GEMM  M=8192 N=4096 K=64  (run twice: stats pass
//                            + fused BN/residual pass; 'pre' never stored)
//
// GEMM staging uses TENSOR_LOAD_TO_LDS (TDM DMA) with double buffering:
// one descriptor per 64x64 fp32 tile, issued by wave 0, gated with
// s_wait_tensorcnt, overlapped with WMMA compute on the other buffer.
// fp32 -> bf16 conversion at fragment build uses the native cast so the
// backend emits v_cvt_pk_bf16_f32 (RNE), co-executing with WMMA.
// ---------------------------------------------------------------------------

typedef __attribute__((ext_vector_type(16))) __bf16 bf16x16;
typedef __attribute__((ext_vector_type(8)))  float  f32x8;
typedef __attribute__((ext_vector_type(4)))  unsigned int u32x4;
typedef __attribute__((ext_vector_type(8)))  int    i32x8;
typedef __attribute__((ext_vector_type(4)))  int    i32x4;

#if defined(__has_builtin)
#  if __has_builtin(__builtin_amdgcn_tensor_load_to_lds)
#    define HAVE_TDM 1
#  endif
#endif
#ifndef HAVE_TDM
#  define HAVE_TDM 0
#endif

__device__ __forceinline__ void put4(bf16x16& d, int base, float4 f) {
    // native fptrunc -> v_cvt_pk_bf16_f32 (round-to-nearest-even)
    d[base + 0] = (__bf16)f.x; d[base + 1] = (__bf16)f.y;
    d[base + 2] = (__bf16)f.z; d[base + 3] = (__bf16)f.w;
}

__device__ __forceinline__ unsigned lds_off_u32(const void* p) {
    return (unsigned)(unsigned long long)(size_t)p;
}

#if HAVE_TDM
// DMA one 64x64 fp32 tile (row stride = stride_elems) from gsrc into LDS.
// D# packing per CDNA5 ISA (2-D, groups 2/3 = 0).
// This toolchain exposes the 6-arg builtin:
//   (uint32x4 g0, int32x8 g1, int32x4 g2, int32x4 g3, int32x8 extra, i32 cpol)
__device__ __forceinline__ void tdm_load_tile64(const float* gsrc,
                                                unsigned lds_byte_off,
                                                int stride_elems)
{
    unsigned long long ga = (unsigned long long)(size_t)gsrc;
    u32x4 g0;
    g0[0] = 1u;                                        // count=1, user load
    g0[1] = lds_byte_off;                              // lds_addr [63:32]
    g0[2] = (unsigned)ga;                              // global_addr [95:64]
    g0[3] = (unsigned)((ga >> 32) & 0x01FFFFFFull)     // global_addr hi
            | (2u << 30);                              // type=2 ("image")
    unsigned sd = (unsigned)stride_elems;
    i32x8 g1;
    g1[0] = (int)(2u << 16);                           // wg_mask=0, data_size=4B
    g1[1] = (int)((sd & 0xFFFFu) << 16);               // tensor_dim0[15:0]
    g1[2] = (int)(((sd >> 16) & 0xFFFFu)               // tensor_dim0[31:16]
                  | (0xFFFFu << 16));                  // tensor_dim1 lo (big)
    g1[3] = (int)(64u << 16);                          // tensor_dim1 hi=0, tile_dim0=64
    g1[4] = (int)64u;                                  // tile_dim1=64, tile_dim2=0
    g1[5] = (int)sd;                                   // tensor_dim0_stride[31:0]
    g1[6] = 0;                                         // stride hi=0, dim1_stride lo=0
    g1[7] = 0;
    i32x4 z4; z4[0] = 0; z4[1] = 0; z4[2] = 0; z4[3] = 0;
    i32x8 z8;
#pragma unroll
    for (int i = 0; i < 8; ++i) z8[i] = 0;
    __builtin_amdgcn_tensor_load_to_lds(g0, g1, z4, z4, z8, 0);
}
#endif

// C[M,N] = epilogue( A[M,Kc] * W[N,Kc]^T )
// epi: 0 plain | 1 sigmoid(v+bias) | 2 partial col stats (no store)
//      3 x + gamma*(v+bias-mean)*rstd  (residual + batchnorm)
__global__ __launch_bounds__(128)
void gemm_wmma_kernel(const float* __restrict__ A, const float* __restrict__ W,
                      float* __restrict__ C, int M, int N, int Kc, int epi,
                      const float* __restrict__ bias,
                      const float* __restrict__ xres,
                      const float* __restrict__ gamma,
                      const float* __restrict__ mean,
                      const float* __restrict__ rstd,
                      float* __restrict__ psum, float* __restrict__ pssq)
{
    __shared__ float bufA[2][64 * 64];   // 16 KB each: 64 rows x 64 K, fp32
    __shared__ float bufW[2][64 * 64];   // 16 KB each: 64 cols x 64 K, fp32
    // total static LDS = 64 KB; epi==2 reduction scratch aliases bufA below.

    const int tid  = threadIdx.x;
    const int wave = tid >> 5;
    const int lane = tid & 31;
    const int n    = lane & 15;          // col (and row) within 16-tile
    const int h    = lane >> 4;          // half-wave id
    const int row0 = blockIdx.x * 64;
    const int col0 = blockIdx.y * 64;
    (void)M;

    f32x8 acc[4];
#pragma unroll
    for (int ct = 0; ct < 4; ++ct) acc[ct] = (f32x8)0.0f;

    const int nk = Kc >> 6;
    const float* Abase = A + (size_t)row0 * Kc;
    const float* Wbase = W + (size_t)col0 * Kc;

#if HAVE_TDM
    if (wave == 0) {                      // prologue: fill buffer 0 via DMA
        tdm_load_tile64(Abase, lds_off_u32(&bufA[0][0]), Kc);
        tdm_load_tile64(Wbase, lds_off_u32(&bufW[0][0]), Kc);
    }
#endif

    int cur = 0;
    for (int kb = 0; kb < nk; ++kb) {
#if HAVE_TDM
        if (wave == 0) {
            if (kb + 1 < nk) {            // prefetch next tiles into other buf
                tdm_load_tile64(Abase + (size_t)(kb + 1) * 64,
                                lds_off_u32(&bufA[cur ^ 1][0]), Kc);
                tdm_load_tile64(Wbase + (size_t)(kb + 1) * 64,
                                lds_off_u32(&bufW[cur ^ 1][0]), Kc);
                __builtin_amdgcn_s_wait_tensorcnt((short)2);  // cur tiles done
            } else {
                __builtin_amdgcn_s_wait_tensorcnt((short)0);
            }
        }
        __syncthreads();                  // publish DMA'd tiles to all waves
#else
        {   // fallback staging: 128 threads copy both 64x64 fp32 tiles
            const int arow = tid >> 1;
            const int koff = (tid & 1) * 32;
            const float4* pa4 = (const float4*)(Abase + (size_t)arow * Kc
                                                + (size_t)kb * 64 + koff);
            const float4* pw4 = (const float4*)(Wbase + (size_t)arow * Kc
                                                + (size_t)kb * 64 + koff);
            float4* da = (float4*)&bufA[cur][arow * 64 + koff];
            float4* dw = (float4*)&bufW[cur][arow * 64 + koff];
#pragma unroll
            for (int q = 0; q < 8; ++q) { da[q] = pa4[q]; dw[q] = pw4[q]; }
        }
        __syncthreads();
#endif

        // ---- 2 x (16x16x32) wmma per col-tile, operands built from LDS ----
        const float* abase = &bufA[cur][(wave * 16 + n) * 64];
#pragma unroll
        for (int kk = 0; kk < 64; kk += 32) {
            bf16x16 afrag;   // lane: row n, K = kk+16h .. kk+16h+15 (contiguous)
            {
                const float4* lo = (const float4*)(abase + kk + h * 16);
                const float4* hi = (const float4*)(abase + kk + h * 16 + 8);
                put4(afrag, 0, lo[0]); put4(afrag, 4,  lo[1]);
                put4(afrag, 8, hi[0]); put4(afrag, 12, hi[1]);
            }
#pragma unroll
            for (int ct = 0; ct < 4; ++ct) {
                // lane: col n, K = kk+16h .. kk+16h+15 (contiguous)
                const float4* bb = (const float4*)
                    (&bufW[cur][(ct * 16 + n) * 64 + kk + h * 16]);
                bf16x16 bfrag;
                put4(bfrag, 0, bb[0]); put4(bfrag, 4,  bb[1]);
                put4(bfrag, 8, bb[2]); put4(bfrag, 12, bb[3]);
                acc[ct] = __builtin_amdgcn_wmma_f32_16x16x32_bf16(
                    false, afrag, false, bfrag, (short)0, acc[ct],
                    false, false);
            }
        }
        __syncthreads();                  // tiles consumed; buffer reusable
        cur ^= 1;
    }

    // ---- epilogue: lane holds rows rowbase..rowbase+7, col = col0+16ct+n --
    const int rowbase = row0 + wave * 16 + h * 8;

    if (epi == 0) {
#pragma unroll
        for (int ct = 0; ct < 4; ++ct) {
            const int col = col0 + ct * 16 + n;
#pragma unroll
            for (int r = 0; r < 8; ++r)
                C[(size_t)(rowbase + r) * N + col] = acc[ct][r];
        }
    } else if (epi == 1) {
#pragma unroll
        for (int ct = 0; ct < 4; ++ct) {
            const int col = col0 + ct * 16 + n;
            const float bv = bias[col];
#pragma unroll
            for (int r = 0; r < 8; ++r) {
                float v = acc[ct][r] + bv;
                C[(size_t)(rowbase + r) * N + col] = 1.0f / (1.0f + __expf(-v));
            }
        }
    } else if (epi == 2) {
        // reuse bufA LDS as reduction scratch (last barrier already passed)
        float* red_s = &bufA[0][0];        // [8][64]
        float* red_q = &bufA[0][0] + 512;  // [8][64]
#pragma unroll
        for (int ct = 0; ct < 4; ++ct) {
            const int col = col0 + ct * 16 + n;
            const float bv = bias[col];
            float S = 0.0f, Q = 0.0f;
#pragma unroll
            for (int r = 0; r < 8; ++r) {
                float v = acc[ct][r] + bv;
                S += v; Q += v * v;
            }
            red_s[(wave * 2 + h) * 64 + ct * 16 + n] = S;
            red_q[(wave * 2 + h) * 64 + ct * 16 + n] = Q;
        }
        __syncthreads();
        if (tid < 64) {
            float S = 0.0f, Q = 0.0f;
#pragma unroll
            for (int i = 0; i < 8; ++i) {
                S += red_s[i * 64 + tid];
                Q += red_q[i * 64 + tid];
            }
            psum[(size_t)blockIdx.x * N + col0 + tid] = S;
            pssq[(size_t)blockIdx.x * N + col0 + tid] = Q;
        }
    } else { // epi == 3: batchnorm + residual
#pragma unroll
        for (int ct = 0; ct < 4; ++ct) {
            const int col = col0 + ct * 16 + n;
            const float bv = bias[col];
            const float mu = mean[col];
            const float rs = rstd[col];
            const float g  = gamma[col];
#pragma unroll
            for (int r = 0; r < 8; ++r) {
                const size_t idx = (size_t)(rowbase + r) * N + col;
                float v = acc[ct][r] + bv;
                C[idx] = xres[idx] + g * (v - mu) * rs;
            }
        }
    }
}

// y[b,:] = H[b] @ v[b,:]  — per-example 64x64 matvec; memory bound on H.
__global__ __launch_bounds__(256)
void hyper_apply_kernel(const float* __restrict__ H, const float* __restrict__ v,
                        float* __restrict__ y)
{
    __shared__ float vsh[4][64];
    const int tid = threadIdx.x;
    vsh[tid >> 6][tid & 63] = v[(size_t)blockIdx.x * 256 + tid];
    __syncthreads();

    const int r = tid >> 6;          // 0..3
    const int k = tid & 63;          // 0..63
    const size_t b = (size_t)blockIdx.x * 4 + r;

    const float4* hr = (const float4*)(H + (b * 64 + k) * 64);
    float sum = 0.0f;
#pragma unroll
    for (int j = 0; j < 16; ++j) {
        float4 hv = hr[j];
        sum += hv.x * vsh[r][4*j+0] + hv.y * vsh[r][4*j+1]
             + hv.z * vsh[r][4*j+2] + hv.w * vsh[r][4*j+3];
    }
    y[b * 64 + k] = sum;
}

// Deterministic final reduction of per-rowblock partials -> mean, rstd.
__global__ __launch_bounds__(256)
void stats_final_kernel(const float* __restrict__ ps, const float* __restrict__ pq,
                        float* __restrict__ mn, float* __restrict__ rs,
                        int nblk, int N, int Btot)
{
    const int c = blockIdx.x * blockDim.x + threadIdx.x;
    if (c >= N) return;
    float S = 0.0f, Q = 0.0f;
    for (int i = 0; i < nblk; ++i) {
        S += ps[(size_t)i * N + c];
        Q += pq[(size_t)i * N + c];
    }
    const float m   = S / (float)Btot;
    const float var = (Q - (float)Btot * m * m) / (float)(Btot - 1);
    mn[c] = m;
    rs[c] = rsqrtf(var + 1e-5f);
}

extern "C" void kernel_launch(void* const* d_in, const int* in_sizes, int n_in,
                              void* d_out, int out_size, void* d_ws, size_t ws_size,
                              hipStream_t stream)
{
    (void)in_sizes; (void)n_in; (void)out_size; (void)ws_size;
    const float* x  = (const float*)d_in[0];   // [B,D]
    const float* H  = (const float*)d_in[1];   // [B,K,K]
    const float* Ld = (const float*)d_in[2];   // [BOT,K]
    const float* Rd = (const float*)d_in[3];   // [K,D]
    const float* Lu = (const float*)d_in[4];   // [D,K]
    const float* Ru = (const float*)d_in[5];   // [K,BOT]
    const float* b1 = (const float*)d_in[6];   // [BOT]
    const float* b2 = (const float*)d_in[7];   // [D]
    const float* gm = (const float*)d_in[8];   // [D]
    float* out = (float*)d_out;                // [B,D]

    const int B = 8192, D = 4096, BOT = 1024, K = 64;

    float* ws = (float*)d_ws;
    float* t   = ws;                               // [B,K]
    float* t2  = t   + (size_t)B * K;              // [B,K]
    float* s   = t2  + (size_t)B * K;              // [B,K]
    float* s2  = s   + (size_t)B * K;              // [B,K]
    float* z   = s2  + (size_t)B * K;              // [B,BOT]
    float* psS = z   + (size_t)B * BOT;            // [B/64, D]
    float* psQ = psS + (size_t)(B / 64) * D;       // [B/64, D]
    float* mn  = psQ + (size_t)(B / 64) * D;       // [D]
    float* rs  = mn  + D;                          // [D]

    const dim3 blk(128);

    // 1) t = x @ Rd^T
    gemm_wmma_kernel<<<dim3(B/64, K/64), blk, 0, stream>>>(
        x, Rd, t, B, K, D, 0, nullptr, nullptr, nullptr, nullptr, nullptr,
        nullptr, nullptr);
    // 2) t2 = H @ t
    hyper_apply_kernel<<<B/4, 256, 0, stream>>>(H, t, t2);
    // 3) z = sigmoid(t2 @ Ld^T + b1)
    gemm_wmma_kernel<<<dim3(B/64, BOT/64), blk, 0, stream>>>(
        t2, Ld, z, B, BOT, K, 1, b1, nullptr, nullptr, nullptr, nullptr,
        nullptr, nullptr);
    // 4) s = z @ Ru^T
    gemm_wmma_kernel<<<dim3(B/64, K/64), blk, 0, stream>>>(
        z, Ru, s, B, K, BOT, 0, nullptr, nullptr, nullptr, nullptr, nullptr,
        nullptr, nullptr);
    // 5) s2 = H @ s
    hyper_apply_kernel<<<B/4, 256, 0, stream>>>(H, s, s2);
    // 6) column stats of (s2 @ Lu^T + b2)  (no 128 MB materialization)
    gemm_wmma_kernel<<<dim3(B/64, D/64), blk, 0, stream>>>(
        s2, Lu, nullptr, B, D, K, 2, b2, nullptr, nullptr, nullptr, nullptr,
        psS, psQ);
    // 7) mean / rstd
    stats_final_kernel<<<D/256, 256, 0, stream>>>(psS, psQ, mn, rs, B/64, D, B);
    // 8) out = x + gamma * ((s2 @ Lu^T + b2) - mean) * rstd   (recompute GEMM)
    gemm_wmma_kernel<<<dim3(B/64, D/64), blk, 0, stream>>>(
        s2, Lu, out, B, D, K, 3, b2, x, gm, mn, rs, nullptr, nullptr);
}
